// MultiheadAttention_23218593202738
// MI455X (gfx1250) — matrix operand order repaired and checked by reading, not compiled, not run on hardware
//
#include <hip/hip_runtime.h>
#include <stdint.h>

// Problem constants (reference: B=4, S=2048, D=512, H=8, dh=64)
#define B_  4
#define S_  2048
#define D_  512
#define H_  8
#define DH_ 64

#define SCALE_   0.125f        /* 1/sqrt(64), exact power of two */
#define EPSILON_ (-50000.0f)

// LDS row paddings in bf16 elements (row strides stay multiples of 8 bf16 = 16B)
#define KPAD 72                // K tile rows: 32 keys x 64 dh
#define VPAD 40                // V^T tile rows: 64 dh x 32 keys
#define WPAD 520               // W^T rows: 16 n x 512 k

typedef __attribute__((ext_vector_type(16))) __bf16 v16bf;
typedef __attribute__((ext_vector_type(8)))  float  v8f;
typedef __attribute__((ext_vector_type(4)))  int    v4i;

typedef __attribute__((address_space(1))) v4i* gv4i_p;   // global int4*
typedef __attribute__((address_space(3))) v4i* lv4i_p;   // LDS int4*

__device__ __forceinline__ __bf16 tobf(float x) { return (__bf16)x; }

// -------- CDNA5 async global->LDS copy (builtin confirmed present) ----------
#if defined(__has_builtin)
#if __has_builtin(__builtin_amdgcn_global_load_async_to_lds_b128) && \
    __has_builtin(__builtin_amdgcn_s_wait_asynccnt)
#define USE_ASYNC_LDS 1
#endif
#endif

__device__ __forceinline__ void cp16_to_lds(__bf16* lds_dst, const __bf16* gsrc) {
#ifdef USE_ASYNC_LDS
  __builtin_amdgcn_global_load_async_to_lds_b128(
      (gv4i_p)(const void*)gsrc, (lv4i_p)(void*)lds_dst, 0, 0);
#else
  *(uint4*)lds_dst = *(const uint4*)gsrc;
#endif
}
__device__ __forceinline__ void wait_async_copies() {
#ifdef USE_ASYNC_LDS
  __builtin_amdgcn_s_wait_asynccnt(0);
#endif
}

// -------- fragment loaders -------------------------------------------------
// A fragment (contiguous k = 2*koff + s) from f32, pre-scaled: lane<16 koff=0, >=16 koff=8.
__device__ __forceinline__ v16bf load_a_f32_scaled(const float* __restrict__ p,
                                                   int koff, float scale) {
  v16bf r;
#pragma unroll
  for (int j = 0; j < 8; ++j) r[j] = tobf(p[2 * koff + j] * scale);
#pragma unroll
  for (int j = 0; j < 8; ++j) r[8 + j] = tobf(p[2 * koff + 8 + j] * scale);
  return r;
}
// A fragment from packed bf16 (two 16B chunks: [2*koff..+7], [2*koff+8..+7]).
__device__ __forceinline__ v16bf load_a_bf16(const __bf16* p, int koff) {
  union { uint4 u[2]; v16bf v; } u;
  u.u[0] = *(const uint4*)(p + 2 * koff);
  u.u[1] = *(const uint4*)(p + 2 * koff + 8);
  return u.v;
}
// B fragment: 16 consecutive packed bf16 (two 16B chunks).
__device__ __forceinline__ v16bf load_b_bf16(const __bf16* p) {
  union { uint4 u[2]; v16bf v; } u;
  u.u[0] = *(const uint4*)(p);
  u.u[1] = *(const uint4*)(p + 8);
  return u.v;
}
// Constant all-ones B fragment (for row-sum-via-WMMA).
__device__ __forceinline__ v16bf ones_frag() {
  v16bf r;
#pragma unroll
  for (int j = 0; j < 16; ++j) r[j] = (__bf16)1.0f;
  return r;
}

// Max across all 32 lanes (whole 16-row score tile shares one max).
__device__ __forceinline__ float allmax32(float v) {
#pragma unroll
  for (int m = 1; m <= 16; m <<= 1) v = fmaxf(v, __shfl_xor(v, m, 32));
  return v;
}

// ---------------------------------------------------------------------------
// Pre-pass 1: K [B,S,H*64] f32  ->  Kbf [B,H,S,64] bf16 (head-contiguous rows)
// ---------------------------------------------------------------------------
__global__ __launch_bounds__(256) void cvt_k_kernel(const float* __restrict__ in,
                                                    __bf16* __restrict__ out) {
  const int idx = blockIdx.x * 256 + threadIdx.x;      // 524288 threads, 8 elems each
  const int d8 = (idx & 7) << 3;
  const int s  = (idx >> 3) & (S_ - 1);
  const int bh = idx >> 14;                            // b*H+h
  const int b = bh >> 3, h = bh & 7;
  const float* src = in + (size_t)(b * S_ + s) * D_ + h * DH_ + d8;
  const float4 a = *(const float4*)(src);
  const float4 c = *(const float4*)(src + 4);
  union { __bf16 b[8]; uint4 u; } pk;
  pk.b[0] = tobf(a.x); pk.b[1] = tobf(a.y); pk.b[2] = tobf(a.z); pk.b[3] = tobf(a.w);
  pk.b[4] = tobf(c.x); pk.b[5] = tobf(c.y); pk.b[6] = tobf(c.z); pk.b[7] = tobf(c.w);
  *(uint4*)(out + (size_t)idx * 8) = pk.u;
}

// ---------------------------------------------------------------------------
// Pre-pass 2: V [B,S,H*64] f32 -> Vbf [B,H,64,S] bf16, transposed and
// pre-scaled by the per-key post-softmax cutoff (cutoff folded into V).
// ---------------------------------------------------------------------------
__global__ __launch_bounds__(256) void cvt_v_kernel(const float* __restrict__ in,
                                                    const float* __restrict__ cutoff,
                                                    __bf16* __restrict__ out) {
  const int idx = blockIdx.x * 256 + threadIdx.x;      // 524288 threads, 8 elems each
  const int s8 = (idx & 255) << 3;                     // 8 consecutive s
  const int d  = (idx >> 8) & 63;
  const int bh = idx >> 14;
  const int b = bh >> 3, h = bh & 7;
  const float* src = in + (size_t)(b * S_ + s8) * D_ + h * DH_ + d;
  const float* cut = cutoff + b * S_ + s8;
  union { __bf16 b[8]; uint4 u; } pk;
#pragma unroll
  for (int j = 0; j < 8; ++j) pk.b[j] = tobf(src[(size_t)j * D_] * cut[j]);
  *(uint4*)(out + (size_t)idx * 8) = pk.u;
}

// ---------------------------------------------------------------------------
// Kernel: fused flash attention. Block = 128 threads (4 wave32) per
// (b, h, 64-query slab); double-buffered bf16 K/V tiles in LDS (async copies).
// Row sums tracked by an extra WMMA against an all-ones B fragment; one
// wave-uniform shared max per 16-row tile (scalar-branch rescale).
// ---------------------------------------------------------------------------
__global__ __launch_bounds__(128) void attn_flash_wmma(
    const float* __restrict__ Q, const __bf16* __restrict__ Kbf,
    const __bf16* __restrict__ Vbf, const unsigned char* __restrict__ mask,
    __bf16* __restrict__ att) {
  __shared__ __bf16 Kl[2][32 * KPAD];   // [key][dh]
  __shared__ __bf16 Vl[2][64 * VPAD];   // [dh][key] (cutoff-scaled)
  __shared__ __bf16 Pl[4][16 * 32];     // per-wave P staging [qrow][key]

  const int tid  = threadIdx.x;
  const int lane = tid & 31;
  const int wave = tid >> 5;

  const int qb = blockIdx.x & 31;       // 32 query slabs of 64
  const int bh = blockIdx.x >> 5;       // b*H+h
  const int b  = bh >> 3;
  const int h  = bh & 7;

  const int row16  = lane & 15;
  const int half   = lane >> 4;
  const int koffA  = half ? 8 : 0;
  const int base16 = half ? 16 : 0;

  const __bf16* Kg = Kbf + (size_t)bh * S_ * DH_;   // [S][64] rows contiguous
  const __bf16* Vg = Vbf + (size_t)bh * DH_ * S_;   // [64][S] rows contiguous
  const unsigned char* mb = mask + b * S_;

  // staging chunk assignment: 256 16B-chunks per tile, 2 per thread
  const int kc_key = tid >> 3, kc_c8 = tid & 7;     // K: 8 chunks per 64-elem row
  const int vc_row = tid >> 2, vc_q  = tid & 3;     // V: 4 chunks per 32-elem row

#define STAGE_TILE(buf, kb)                                                       \
  {                                                                               \
    _Pragma("unroll") for (int i = 0; i < 2; ++i) {                               \
      const int kk = kc_key + i * 16;                                             \
      cp16_to_lds(&Kl[buf][kk * KPAD + kc_c8 * 8],                                \
                  Kg + (size_t)((kb) + kk) * DH_ + kc_c8 * 8);                    \
      const int vr = vc_row + i * 32;                                             \
      cp16_to_lds(&Vl[buf][vr * VPAD + vc_q * 8],                                 \
                  Vg + (size_t)vr * S_ + (kb) + vc_q * 8);                        \
    }                                                                             \
  }

  // Q fragments, pre-scaled by 1/sqrt(dh) (power of two: exact in bf16).
  const int qrow = qb * 64 + wave * 16 + row16;
  const float* qptr = Q + (size_t)(b * S_ + qrow) * D_ + h * DH_;
  const v16bf aq0 = load_a_f32_scaled(qptr, koffA, SCALE_);
  const v16bf aq1 = load_a_f32_scaled(qptr + 32, koffA, SCALE_);
  const v16bf bones = ones_frag();

  v8f acc[4] = {};
  v8f accS = {};                        // running row-sum (Z), via WMMA
  float rmax = -INFINITY;               // wave-uniform shared running max

  STAGE_TILE(0, 0);
  int buf = 0;

  for (int kb = 0; kb < S_; kb += 32) {
    wait_async_copies();
    __syncthreads();                       // tile `buf` ready, prev compute done
    if (kb + 32 < S_) STAGE_TILE(buf ^ 1, kb + 32);

    const __bf16* kl = &Kl[buf][0];
    const __bf16* vl = &Vl[buf][0];

    // Scores: (Q/8)(16x64) x K^T(64x32) -> two 16x16 half-tiles.
    const v16bf bk00 = load_b_bf16(kl + row16 * KPAD + base16);
    const v16bf bk01 = load_b_bf16(kl + row16 * KPAD + 32 + base16);
    const v16bf bk10 = load_b_bf16(kl + (16 + row16) * KPAD + base16);
    const v16bf bk11 = load_b_bf16(kl + (16 + row16) * KPAD + 32 + base16);

    v8f sc0 = {}, sc1 = {};
    sc0 = __builtin_amdgcn_wmma_f32_16x16x32_bf16(false, aq0, false, bk00, (short)0, sc0, false, false);
    sc0 = __builtin_amdgcn_wmma_f32_16x16x32_bf16(false, aq1, false, bk01, (short)0, sc0, false, false);
    sc1 = __builtin_amdgcn_wmma_f32_16x16x32_bf16(false, aq0, false, bk10, (short)0, sc1, false, false);
    sc1 = __builtin_amdgcn_wmma_f32_16x16x32_bf16(false, aq1, false, bk11, (short)0, sc1, false, false);

    const bool m0 = mb[kb + row16] != 0;
    const bool m1 = mb[kb + 16 + row16] != 0;

    float s0a[8], s1a[8];
    float tmax = -INFINITY;
#pragma unroll
    for (int r = 0; r < 8; ++r) {
      s0a[r] = m0 ? sc0[r] : EPSILON_;
      s1a[r] = m1 ? sc1[r] : EPSILON_;
      tmax = fmaxf(tmax, fmaxf(s0a[r], s1a[r]));
    }
    tmax = allmax32(tmax);                 // shared max, uniform across wave

    const float nmax = fmaxf(rmax, tmax);
    if (nmax > rmax) {                     // wave-uniform -> scalar branch
      const float alpha = __expf(rmax - nmax);
#pragma unroll
      for (int r = 0; r < 8; ++r) {
        acc[0][r] *= alpha; acc[1][r] *= alpha;
        acc[2][r] *= alpha; acc[3][r] *= alpha;
        accS[r]   *= alpha;
      }
    }
    rmax = nmax;

    // P = exp(s - max): C-layout regs -> A-layout via per-wave bf16 LDS tile.
    __bf16* pw = &Pl[wave][0];
#pragma unroll
    for (int r = 0; r < 8; ++r) {
      const int pr = r + (half ? 8 : 0);
      pw[pr * 32 + row16]      = tobf(__expf(s0a[r] - nmax));
      pw[pr * 32 + 16 + row16] = tobf(__expf(s1a[r] - nmax));
    }
    __builtin_amdgcn_wave_barrier();

    const v16bf ap = load_a_bf16(pw + row16 * 32, koffA);
#pragma unroll
    for (int t = 0; t < 4; ++t) {
      const v16bf bv = load_b_bf16(vl + (t * 16 + row16) * VPAD + base16);
      acc[t] = __builtin_amdgcn_wmma_f32_16x16x32_bf16(false, ap, false, bv, (short)0, acc[t], false, false);
    }
    // Row sums: one WMMA against ones replaces 32 cross-lane reductions.
    accS = __builtin_amdgcn_wmma_f32_16x16x32_bf16(false, ap, false, bones, (short)0, accS, false, false);

    buf ^= 1;
  }

  // Epilogue: O / Z (Z replicated in every lane via C layout), store bf16.
  float inv[8];
#pragma unroll
  for (int r = 0; r < 8; ++r) inv[r] = 1.0f / accS[r];
#pragma unroll
  for (int t = 0; t < 4; ++t) {
#pragma unroll
    for (int r = 0; r < 8; ++r) {
      const int orow = qb * 64 + wave * 16 + r + (half ? 8 : 0);
      att[(size_t)(b * S_ + orow) * D_ + h * DH_ + t * 16 + row16] =
          tobf(acc[t][r] * inv[r]);
    }
  }
#undef STAGE_TILE
}

// ---------------------------------------------------------------------------
// Kernel: out = att[8192x512](bf16) @ W[512x512](f32)  via bf16 WMMA.
// Block = 4 waves share one 16-wide W column slab transposed in LDS (bf16).
// ---------------------------------------------------------------------------
__global__ __launch_bounds__(128) void proj_wmma(
    const __bf16* __restrict__ A, const float* __restrict__ W,
    float* __restrict__ out) {
  __shared__ __bf16 Wl[16 * WPAD];      // W^T slab: [n(16)][k(512)]

  const int tid  = threadIdx.x;
  const int lane = tid & 31;
  const int wave = tid >> 5;
  const int nb = blockIdx.x & 31;
  const int mg = blockIdx.x >> 5;

  const int row16  = lane & 15;
  const int half   = lane >> 4;
  const int koffA  = half ? 8 : 0;
  const int base16 = half ? 16 : 0;

#pragma unroll
  for (int i = 0; i < 16; ++i) {
    const int e4 = tid + i * 128;
    const int k  = e4 >> 2;
    const int c4 = e4 & 3;
    const float4 wv = *(const float4*)(W + (size_t)k * D_ + nb * 16 + c4 * 4);
    Wl[(c4 * 4 + 0) * WPAD + k] = tobf(wv.x);
    Wl[(c4 * 4 + 1) * WPAD + k] = tobf(wv.y);
    Wl[(c4 * 4 + 2) * WPAD + k] = tobf(wv.z);
    Wl[(c4 * 4 + 3) * WPAD + k] = tobf(wv.w);
  }
  __syncthreads();

  const int mrow = mg * 64 + wave * 16 + row16;
  const __bf16* arow = A + (size_t)mrow * D_;

  v8f acc = {};
#pragma unroll
  for (int k0 = 0; k0 < D_; k0 += 32) {
    const v16bf af = load_a_bf16(arow + k0, koffA);
    const v16bf bf = load_b_bf16(&Wl[row16 * WPAD + k0 + base16]);
    acc = __builtin_amdgcn_wmma_f32_16x16x32_bf16(false, af, false, bf, (short)0, acc, false, false);
  }

#pragma unroll
  for (int r = 0; r < 8; ++r) {
    const int row = mg * 64 + wave * 16 + r + (half ? 8 : 0);
    out[(size_t)row * D_ + nb * 16 + row16] = acc[r];
  }
}

extern "C" void kernel_launch(void* const* d_in, const int* in_sizes, int n_in,
                              void* d_out, int out_size, void* d_ws, size_t ws_size,
                              hipStream_t stream) {
  const float*         q      = (const float*)d_in[0];
  const float*         k      = (const float*)d_in[1];
  const float*         v      = (const float*)d_in[2];
  const unsigned char* mask   = (const unsigned char*)d_in[3];  // bool -> u8
  const float*         cutoff = (const float*)d_in[4];
  const float*         W      = (const float*)d_in[5];
  float* out = (float*)d_out;

  // workspace layout (bf16): Kbf 8MiB | Vbf 8MiB | att 8MiB
  __bf16* kbf = (__bf16*)d_ws;
  __bf16* vbf = kbf + (size_t)B_ * H_ * S_ * DH_;
  __bf16* att = vbf + (size_t)B_ * H_ * S_ * DH_;

  cvt_k_kernel<<<dim3(2048), dim3(256), 0, stream>>>(k, kbf);
  cvt_v_kernel<<<dim3(2048), dim3(256), 0, stream>>>(v, cutoff, vbf);
  attn_flash_wmma<<<dim3(1024), dim3(128), 0, stream>>>(q, kbf, vbf, mask, att);
  proj_wmma<<<dim3(4096), dim3(128), 0, stream>>>(att, W, out);
}
